// CrossMultiheadSelfAttention_66468913873567
// MI455X (gfx1250) — matrix operand order repaired and checked by reading, not compiled, not run on hardware
//
#include <hip/hip_runtime.h>
#include <hip/hip_bf16.h>

typedef _Float16 v8h  __attribute__((ext_vector_type(8)));
typedef _Float16 v16h __attribute__((ext_vector_type(16)));
typedef float    v8f  __attribute__((ext_vector_type(8)));

#define BATCH 4
#define CCH   128
#define HH    256
#define WW    256
#define NPIX  65536ULL
#define HEADS 8
#define TPAD  144  // LDS tile row stride in halfs: 288B rows -> bank spread, 16B aligned

// ---------------- WMMA fragment loaders (wave32, 16x16x32 f16) ----------------
// A-matrix 16x32: lane L (0-15) row M=L, K chunks {0..7,16..23}; lanes 16-31 row M=L-16,
// K chunks {8..15,24..31}.  A stored row-major (M x ld).
__device__ __forceinline__ v16h load_frag_a(const _Float16* __restrict__ A, size_t ld,
                                            int m0, size_t k0, int lane) {
  const _Float16* p = A + (size_t)(m0 + (lane & 15)) * ld + k0 + (size_t)((lane >> 4) << 3);
  v16h a;
  *(v8h*)&a         = *(const v8h*)p;        // K = k0 + hk .. +7
  *(((v8h*)&a) + 1) = *(const v8h*)(p + 16); // K = k0 + 16 + hk .. +7
  return a;
}
// B-matrix 32x16: lane L column N = L&15; lanes 0-15 hold K {0..7,16..23}, lanes 16-31
// K {8..15,24..31}.  Bt stored as (col, K) row-major with leading dim ld.
__device__ __forceinline__ v16h load_frag_b(const _Float16* __restrict__ Bt, size_t ld,
                                            int col0, size_t k0, int lane) {
  const _Float16* p = Bt + (size_t)(col0 + (lane & 15)) * ld + k0 + (size_t)((lane >> 4) << 3);
  v16h b;
  *(v8h*)&b         = *(const v8h*)p;       // K = k0+kb .. +7
  *(((v8h*)&b) + 1) = *(const v8h*)(p + 16); // K = k0+16+kb .. +7
  return b;
}

// ---------------- 1. LayerNorm over channels, write f16 pixel-major ----------------
__global__ void ln_transpose_kernel(const float* __restrict__ x, const float* __restrict__ w,
                                    const float* __restrict__ bias, _Float16* __restrict__ outT) {
  int b = blockIdx.y;
  size_t n = (size_t)blockIdx.x * blockDim.x + threadIdx.x;
  const float* xb = x + (size_t)b * CCH * NPIX;
  float s = 0.f, ss = 0.f;
  for (int c = 0; c < CCH; ++c) {
    float v = xb[(size_t)c * NPIX + n];
    s += v; ss += v * v;
  }
  float mu = s * (1.0f / CCH);
  float var = ss * (1.0f / CCH) - mu * mu;
  float inv = rsqrtf(var + 1e-5f);
  _Float16* o = outT + ((size_t)b * NPIX + n) * CCH;
  for (int c = 0; c < CCH; ++c) {
    float v = xb[(size_t)c * NPIX + n];
    o[c] = (_Float16)((v - mu) * inv * w[c] + bias[c]);
  }
}

// ---------------- 2. f32 -> f16 cast ----------------
__global__ void cvt_f16_kernel(const float* __restrict__ in, _Float16* __restrict__ out, int n) {
  int i = blockIdx.x * blockDim.x + threadIdx.x;
  if (i < n) out[i] = (_Float16)in[i];
}

// ---------------- 3. conv1x1 / final projection as WMMA GEMM with async-LDS staging --------
// Y[b,m,n] = A[m,:] . XT[b,n,:], K = 128.  Block = 8 waves covering 128 M-rows x 64 pixels.
// The 64x128 f16 activation tile (16KB) is staged to LDS ONCE per block with
// global_load_async_to_lds_b128 (ASYNCcnt), then all 8 waves read B-fragments from LDS,
// removing the 8x redundant global traffic of the naive version.
template <typename OutT>
__global__ void gemm_wmma_staged_kernel(const _Float16* __restrict__ Am,
                                        const _Float16* __restrict__ XT,
                                        OutT* __restrict__ Y, int Mtot, int perBatchA) {
  __shared__ __align__(16) _Float16 tile[64 * TPAD];
  int b = blockIdx.z;
  int lane = threadIdx.x & 31, wave = threadIdx.x >> 5;
  int m0 = blockIdx.y * 128 + wave * 16;
  size_t n0 = (size_t)blockIdx.x * 64;
  const _Float16* A = Am + (perBatchA ? (size_t)b * (size_t)Mtot * CCH : 0);
  const _Float16* xt = XT + (size_t)b * NPIX * CCH;
  OutT* y = Y + (size_t)b * (size_t)Mtot * NPIX;

  // ---- async copy: 64 rows x 256B -> LDS (each thread 4x b128) ----
#pragma unroll
  for (int i = 0; i < 4; ++i) {
    int j = i * 256 + (int)threadIdx.x;
    int row = j >> 4;
    int ch  = (j & 15) << 3;
    unsigned long long ga =
        (unsigned long long)(uintptr_t)(xt + (n0 + (size_t)row) * CCH + ch);
    unsigned int la = (unsigned int)(uintptr_t)(&tile[row * TPAD + ch]);
    asm volatile("global_load_async_to_lds_b128 %0, %1, off" ::"v"(la), "v"(ga) : "memory");
  }
  asm volatile("s_wait_asynccnt 0" ::: "memory");
  __syncthreads();

  v8f acc[4] = {};
#pragma unroll
  for (int k0 = 0; k0 < 128; k0 += 32) {
    v16h a = load_frag_a(A, CCH, m0, (size_t)k0, lane);
#pragma unroll
    for (int j = 0; j < 4; ++j) {
      const _Float16* p = &tile[(j * 16 + (lane & 15)) * TPAD + k0 + ((lane >> 4) << 3)];
      v16h bf;
      *(v8h*)&bf         = *(const v8h*)p;
      *(((v8h*)&bf) + 1) = *(const v8h*)(p + 16);
      acc[j] = __builtin_amdgcn_wmma_f32_16x16x32_f16(false, a, false, bf, (short)0, acc[j],
                                                      false, false);
    }
  }
  int ncol = lane & 15, mb = (lane >> 4) * 8;
#pragma unroll
  for (int j = 0; j < 4; ++j)
#pragma unroll
    for (int r = 0; r < 8; ++r)
      y[(size_t)(m0 + mb + r) * NPIX + n0 + 16 * j + ncol] = (OutT)acc[j][r];
}

// ---------------- 4. depthwise 3x3 conv (f16 in/out, f32 math) ----------------
// transpose_out == 0: Y[b,c,n] channel-major.  ==1: Y[(b*N+n)*128 + c] pixel-major (for v).
__global__ void dwconv3x3_kernel(const _Float16* __restrict__ X, const float* __restrict__ Wdw,
                                 _Float16* __restrict__ Y, int Cstride, int c_base,
                                 int transpose_out) {
  int b = blockIdx.z, c = blockIdx.y;
  int n = blockIdx.x * blockDim.x + threadIdx.x;
  int hy = n >> 8, wx = n & 255;
  const _Float16* plane = X + ((size_t)b * Cstride + c_base + c) * NPIX;
  const float* w9 = Wdw + (size_t)(c_base + c) * 9;
  float acc = 0.f;
#pragma unroll
  for (int dy = 0; dy < 3; ++dy)
#pragma unroll
    for (int dx = 0; dx < 3; ++dx) {
      int yy = hy + dy - 1, xx = wx + dx - 1;
      bool ok = (yy >= 0) && (yy < HH) && (xx >= 0) && (xx < WW);
      float v = ok ? (float)plane[(size_t)yy * WW + xx] : 0.f;
      acc += v * w9[dy * 3 + dx];
    }
  _Float16 r = (_Float16)acc;
  if (!transpose_out)
    Y[((size_t)b * 128 + c) * NPIX + n] = r;
  else
    Y[((size_t)b * NPIX + n) * 128 + c] = r;
}

// ---------------- 5. per-row reciprocal L2 norm over N ----------------
__global__ void row_rnorm_kernel(const _Float16* __restrict__ X, float* __restrict__ scale) {
  size_t base = (size_t)blockIdx.x * NPIX;
  float s = 0.f;
  for (size_t i = threadIdx.x; i < NPIX; i += 256) {
    float v = (float)X[base + i];
    s += v * v;
  }
  __shared__ float sb[256];
  sb[threadIdx.x] = s;
  __syncthreads();
  for (int st = 128; st > 0; st >>= 1) {
    if (threadIdx.x < (unsigned)st) sb[threadIdx.x] += sb[threadIdx.x + st];
    __syncthreads();
  }
  if (threadIdx.x == 0) scale[blockIdx.x] = 1.0f / fmaxf(sqrtf(sb[0]), 1e-12f);
}

// ---------------- 6. attention: attn = softmax(scale * q.kT) per (b,head), K = NPIX ----------
__global__ void attn_wmma_kernel(const _Float16* __restrict__ QP, const _Float16* __restrict__ KP,
                                 const float* __restrict__ qs, const float* __restrict__ ks,
                                 const float* __restrict__ temp, float* __restrict__ attn) {
  int b = blockIdx.x >> 3, hd = blockIdx.x & 7;
  int lane = threadIdx.x & 31, wave = threadIdx.x >> 5;
  size_t hbase = ((size_t)b * CCH + hd * 16) * NPIX;
  const _Float16* qh = QP + hbase;
  const _Float16* kh = KP + hbase;
  v8f acc = {};
  const int chunks = (int)(NPIX / 32); // 2048, strided over 8 waves
  for (int it = wave; it < chunks; it += 8) {
    size_t n0 = (size_t)it * 32;
    v16h a  = load_frag_a(qh, NPIX, 0, n0, lane);
    v16h bf = load_frag_b(kh, NPIX, 0, n0, lane);
    // prefetch this wave's next K-chunk (streaming reduction)
    __builtin_prefetch(qh + (size_t)(lane & 15) * NPIX + n0 + 256, 0, 0);
    __builtin_prefetch(kh + (size_t)(lane & 15) * NPIX + n0 + 256, 0, 0);
    acc = __builtin_amdgcn_wmma_f32_16x16x32_f16(false, a, false, bf, (short)0, acc, false, false);
  }
  __shared__ float red[8][256];
  int ncol = lane & 15, mb = (lane >> 4) * 8;
#pragma unroll
  for (int r = 0; r < 8; ++r) red[wave][(mb + r) * 16 + ncol] = acc[r];
  __syncthreads();
  int idx = threadIdx.x;
  float s = 0.f;
#pragma unroll
  for (int wv = 0; wv < 8; ++wv) s += red[wv][idx];
  int c = idx >> 4, d = idx & 15;
  s *= qs[b * CCH + hd * 16 + c] * ks[b * CCH + hd * 16 + d] * temp[hd];
  __shared__ float mat[256];
  mat[idx] = s;
  __syncthreads();
  if (threadIdx.x < 16) {
    int row = threadIdx.x;
    float mx = -3.4e38f;
    for (int j = 0; j < 16; ++j) mx = fmaxf(mx, mat[row * 16 + j]);
    float e[16], sum = 0.f;
    for (int j = 0; j < 16; ++j) { e[j] = __expf(mat[row * 16 + j] - mx); sum += e[j]; }
    float inv = 1.0f / sum;
    float* ao = attn + (((size_t)(b * HEADS + hd) * 16 + row) * 16);
    for (int j = 0; j < 16; ++j) ao[j] = e[j] * inv;
  }
}

// ---------------- 7. fold Wout into attention: Mb[b][o, hd*16+d] = sum_c Wout[o,hd*16+c]*attn ----
__global__ void fold_wout_kernel(const float* __restrict__ Wout, const float* __restrict__ attn,
                                 _Float16* __restrict__ Mb) {
  int b = blockIdx.x >> 7, o = blockIdx.x & 127;
  int df = threadIdx.x;
  int hd = df >> 4, d = df & 15;
  float s = 0.f;
#pragma unroll
  for (int c = 0; c < 16; ++c)
    s += Wout[o * CCH + hd * 16 + c] * attn[((size_t)(b * HEADS + hd) * 16 + c) * 16 + d];
  Mb[((size_t)b * CCH + o) * CCH + df] = (_Float16)s;
}

extern "C" void kernel_launch(void* const* d_in, const int* in_sizes, int n_in,
                              void* d_out, int out_size, void* d_ws, size_t ws_size,
                              hipStream_t stream) {
  const float* q           = (const float*)d_in[0];
  const float* k           = (const float*)d_in[1];
  const float* ln_q_w      = (const float*)d_in[2];
  const float* ln_q_b      = (const float*)d_in[3];
  const float* ln_k_w      = (const float*)d_in[4];
  const float* ln_k_b      = (const float*)d_in[5];
  const float* Wq          = (const float*)d_in[6];
  const float* Wkv         = (const float*)d_in[7];
  const float* Wq_dw       = (const float*)d_in[8];
  const float* Wkv_dw      = (const float*)d_in[9];
  const float* Wout        = (const float*)d_in[10];
  const float* temperature = (const float*)d_in[11];
  float* out = (float*)d_out;

  char* ws = (char*)d_ws;
  size_t off = 0;
  auto alloc = [&](size_t bytes) -> char* {
    char* p = ws + off;
    off += (bytes + 255) & ~(size_t)255;
    return p;
  };
  _Float16* qnT    = (_Float16*)alloc(BATCH * NPIX * CCH * 2);       // LN(q), pixel-major
  _Float16* knT    = (_Float16*)alloc(BATCH * NPIX * CCH * 2);       // LN(k), pixel-major
  _Float16* Wq16   = (_Float16*)alloc((size_t)CCH * CCH * 2);
  _Float16* Wkv16  = (_Float16*)alloc((size_t)2 * CCH * CCH * 2);
  _Float16* qp_pre = (_Float16*)alloc(BATCH * (size_t)CCH * NPIX * 2);
  _Float16* kv_pre = (_Float16*)alloc(BATCH * (size_t)2 * CCH * NPIX * 2);
  _Float16* qp     = (_Float16*)alloc(BATCH * (size_t)CCH * NPIX * 2); // q after dwconv, (c,n)
  _Float16* kp     = (_Float16*)alloc(BATCH * (size_t)CCH * NPIX * 2); // k after dwconv, (c,n)
  _Float16* vT     = (_Float16*)alloc(BATCH * NPIX * (size_t)CCH * 2); // v after dwconv, (n,c)
  float*    qscale = (float*)alloc(BATCH * CCH * 4);
  float*    kscale = (float*)alloc(BATCH * CCH * 4);
  float*    attn   = (float*)alloc(BATCH * HEADS * 16 * 16 * 4);
  _Float16* Mb     = (_Float16*)alloc(BATCH * (size_t)CCH * CCH * 2);
  (void)ws_size; (void)in_sizes; (void)n_in; (void)out_size;

  // 1. LayerNorm -> f16 pixel-major
  ln_transpose_kernel<<<dim3(NPIX / 256, BATCH), 256, 0, stream>>>(q, ln_q_w, ln_q_b, qnT);
  ln_transpose_kernel<<<dim3(NPIX / 256, BATCH), 256, 0, stream>>>(k, ln_k_w, ln_k_b, knT);

  // 2. cast conv1x1 weights to f16
  cvt_f16_kernel<<<(CCH * CCH + 255) / 256, 256, 0, stream>>>(Wq, Wq16, CCH * CCH);
  cvt_f16_kernel<<<(2 * CCH * CCH + 255) / 256, 256, 0, stream>>>(Wkv, Wkv16, 2 * CCH * CCH);

  // 3. conv1x1 via WMMA (async LDS staging)
  gemm_wmma_staged_kernel<_Float16><<<dim3(NPIX / 64, 1, BATCH), 256, 0, stream>>>(
      Wq16, qnT, qp_pre, CCH, 0);
  gemm_wmma_staged_kernel<_Float16><<<dim3(NPIX / 64, 2, BATCH), 256, 0, stream>>>(
      Wkv16, knT, kv_pre, 2 * CCH, 0);

  // 4. depthwise 3x3
  dwconv3x3_kernel<<<dim3(NPIX / 256, CCH, BATCH), 256, 0, stream>>>(qp_pre, Wq_dw, qp, CCH, 0, 0);
  dwconv3x3_kernel<<<dim3(NPIX / 256, CCH, BATCH), 256, 0, stream>>>(kv_pre, Wkv_dw, kp, 2 * CCH,
                                                                     0, 0);
  dwconv3x3_kernel<<<dim3(NPIX / 256, CCH, BATCH), 256, 0, stream>>>(kv_pre, Wkv_dw, vT, 2 * CCH,
                                                                     CCH, 1);

  // 5. reciprocal L2 row norms
  row_rnorm_kernel<<<BATCH * CCH, 256, 0, stream>>>(qp, qscale);
  row_rnorm_kernel<<<BATCH * CCH, 256, 0, stream>>>(kp, kscale);

  // 6. attention (WMMA over K = NPIX) + softmax
  attn_wmma_kernel<<<BATCH * HEADS, 256, 0, stream>>>(qp, kp, qscale, kscale, temperature, attn);

  // 7. fold Wout * blockdiag(attn) -> per-batch 128x128 f16 matrix
  fold_wout_kernel<<<BATCH * CCH, 128, 0, stream>>>(Wout, attn, Mb);

  // 8. final fused (attn@v + output projection) per-batch GEMM -> f32 output
  gemm_wmma_staged_kernel<float><<<dim3(NPIX / 64, 1, BATCH), 256, 0, stream>>>(
      Mb, vT, out, CCH, 1);
}
